// NystromAttention_72894184947781
// MI455X (gfx1250) — matrix operand order repaired and checked by reading, not compiled, not run on hardware
//
#include <hip/hip_runtime.h>
#include <hip/hip_bf16.h>
#include <math.h>

// ---------------------------------------------------------------------------
// Nystrom attention for MI455X (gfx1250, wave32, WMMA).
// GEMMs: v_wmma_f32_16x16x32_bf16, 64x128 block tile, 8 waves x (32x32),
// double-buffered LDS fed by global_load_async_to_lds_b128 (ASYNCcnt).
// Softmax / Newton-Schulz inverse in fp32 VALU (negligible FLOPs).
// Workspace use: ~103 MB (L2-resident on 192MB L2).
// ---------------------------------------------------------------------------

typedef __attribute__((ext_vector_type(16))) __bf16 v16bf;
typedef __attribute__((ext_vector_type(8)))  __bf16 v8bf;
typedef __attribute__((ext_vector_type(8)))  float  v8f;

#define LDS_STRIDE 72   // 64 + 8 pad (16B) to avoid LDS bank conflicts

// 32-bit LDS offset of a pointer into __shared__ memory (for async-to-LDS ops)
__device__ __forceinline__ unsigned lds_addr_of(const void* p) {
  return (unsigned)(unsigned long long)(const __attribute__((address_space(3))) void*)p;
}

// async copy of 16 bytes global -> LDS (per lane), tracked by ASYNCcnt
__device__ __forceinline__ void async_copy16(unsigned lds_off, const void* gptr) {
  unsigned long long ga = (unsigned long long)(size_t)gptr;
  asm volatile("global_load_async_to_lds_b128 %0, %1, off"
               :: "v"(lds_off), "v"(ga) : "memory");
}

__device__ __forceinline__ void wait_async0() {
  asm volatile("s_wait_asynccnt 0x0" ::: "memory");
}

// A-operand fragment (16x32 bf16, M x K): lane L holds row M=(L&15);
// elements 0..7 -> K = kk + (L>>4)*8 + e ; 8..15 -> K = kk + 16 + (L>>4)*8 + e-8
__device__ __forceinline__ v16bf frag_A(const __bf16* tile, int row0, int kk, int lane) {
  int r  = row0 + (lane & 15);
  int hi = (lane >> 4) * 8;
  const __bf16* p = tile + r * LDS_STRIDE + kk + hi;
  v8bf lo = *(const v8bf*)p;
  v8bf hh = *(const v8bf*)(p + 16);
  v16bf f;
#pragma unroll
  for (int e = 0; e < 8; ++e) { f[e] = lo[e]; f[e + 8] = hh[e]; }
  return f;
}

// B-operand fragment (32x16 bf16, K x N) from an [N][K]-layout tile:
// lane L holds col N=(L&15); elements 0..7 -> K = kk + (L>>4)*8 + e ; 8..15 -> K = kk + 16 + (L>>4)*8 + e-8
__device__ __forceinline__ v16bf frag_B(const __bf16* tile, int row0, int kk, int lane) {
  int r  = row0 + (lane & 15);
  int hi = (lane >> 4) * 8;
  const __bf16* p = tile + r * LDS_STRIDE + kk + hi;
  v8bf lo = *(const v8bf*)p;
  v8bf hh = *(const v8bf*)(p + 16);
  v16bf f;
#pragma unroll
  for (int e = 0; e < 8; ++e) { f[e] = lo[e]; f[e + 8] = hh[e]; }
  return f;
}

// Generic tiled GEMM:  C[M,N] = Aop[M,K] * Bop[K,N]  (+bias[m]) (+addsrc)
//   Aop[m,k] = transA ? A[k*lda+m] : A[m*lda+k]
//   Bop[k,n] = transB ? B[n*ldb+k] : B[k*ldb+n]
// Block tile 64x128xK64; 256 threads = 8 wave32 waves, each wave 32x32.
// M multiple of 64, K multiple of 64; N guarded by Ncols in the epilogue
// (OOB staging reads must land in mapped memory - true for our workspace).
__global__ __launch_bounds__(256) void gemm_bf16_wmma(
    const __bf16* __restrict__ A, int lda, long long strA, int transA,
    const __bf16* __restrict__ B, int ldb, long long strB, int transB,
    float* __restrict__ outF, __bf16* __restrict__ outH, int ldc, long long strC,
    const float* __restrict__ bias,
    const __bf16* __restrict__ addsrc, int ldadd, long long strAdd,
    int K, int Ncols)
{
  __shared__ __bf16 shA[2][64 * LDS_STRIDE];
  __shared__ __bf16 shB[2][128 * LDS_STRIDE];

  const int t    = threadIdx.x;
  const int lane = t & 31;
  const int wid  = t >> 5;
  const int wm   = (wid & 1) * 32;    // wave row base within tile
  const int wn   = (wid >> 1) * 32;   // wave col base within tile
  const int bm0  = blockIdx.y * 64;
  const int bn0  = blockIdx.x * 128;
  const int bz   = blockIdx.z;

  A += (long long)bz * strA;
  B += (long long)bz * strB;

  v8f acc00 = {}, acc01 = {}, acc10 = {}, acc11 = {};

  const int ntiles = K >> 6;

  // stage K-tile `kt` into LDS buffer `buf`
  auto stage = [&](int kt, int buf) {
    __bf16* sA = shA[buf];
    __bf16* sB = shB[buf];
    if (!transA) {
      // contiguous along K: async global->LDS, 16B per lane
      for (int i = t; i < 64 * 8; i += 256) {
        int r = i >> 3, c8 = (i & 7) << 3;
        async_copy16(lds_addr_of(&sA[r * LDS_STRIDE + c8]),
                     A + (size_t)(bm0 + r) * lda + kt + c8);
      }
    } else {
      // transpose via registers: load 8 along M, scatter to 8 LDS rows
      for (int i = t; i < 64 * 8; i += 256) {
        int c = i >> 3, r8 = (i & 7) << 3;
        v8bf v = *(const v8bf*)(A + (size_t)(kt + c) * lda + bm0 + r8);
#pragma unroll
        for (int j = 0; j < 8; ++j) sA[(r8 + j) * LDS_STRIDE + c] = v[j];
      }
    }
    if (transB) {
      for (int i = t; i < 128 * 8; i += 256) {
        int r = i >> 3, c8 = (i & 7) << 3;
        async_copy16(lds_addr_of(&sB[r * LDS_STRIDE + c8]),
                     B + (size_t)(bn0 + r) * ldb + kt + c8);
      }
    } else {
      for (int i = t; i < 64 * 16; i += 256) {
        int c = i >> 4, r8 = (i & 15) << 3;
        v8bf v = *(const v8bf*)(B + (size_t)(kt + c) * ldb + bn0 + r8);
#pragma unroll
        for (int j = 0; j < 8; ++j) sB[(r8 + j) * LDS_STRIDE + c] = v[j];
      }
    }
  };

  stage(0, 0);
  for (int ti = 0; ti < ntiles; ++ti) {
    wait_async0();        // my async writes for tile ti are in LDS
    __syncthreads();      // (also waits DScnt) publish buffers block-wide
    const int cur = ti & 1;
    if (ti + 1 < ntiles) stage((ti + 1) << 6, cur ^ 1);  // prefetch next tile
    const __bf16* cA = shA[cur];
    const __bf16* cB = shB[cur];
#pragma unroll
    for (int kk = 0; kk < 64; kk += 32) {
      v16bf a0 = frag_A(cA, wm,      kk, lane);
      v16bf a1 = frag_A(cA, wm + 16, kk, lane);
      v16bf b0 = frag_B(cB, wn,      kk, lane);
      v16bf b1 = frag_B(cB, wn + 16, kk, lane);
      acc00 = __builtin_amdgcn_wmma_f32_16x16x32_bf16(false, a0, false, b0, (short)0, acc00, false, false);
      acc01 = __builtin_amdgcn_wmma_f32_16x16x32_bf16(false, a0, false, b1, (short)0, acc01, false, false);
      acc10 = __builtin_amdgcn_wmma_f32_16x16x32_bf16(false, a1, false, b0, (short)0, acc10, false, false);
      acc11 = __builtin_amdgcn_wmma_f32_16x16x32_bf16(false, a1, false, b1, (short)0, acc11, false, false);
    }
  }

  // ---- epilogue: C layout N=lane&15, vgpr r -> M = r + 8*(lane>>4) ----
  const int hi = lane >> 4;
  float*  oF = outF ? outF + (long long)bz * strC : nullptr;
  __bf16* oH = outH ? outH + (long long)bz * strC : nullptr;
  const __bf16* av = addsrc ? addsrc + (long long)bz * strAdd : nullptr;
#pragma unroll
  for (int mf = 0; mf < 2; ++mf) {
#pragma unroll
    for (int nf = 0; nf < 2; ++nf) {
      v8f acc = mf ? (nf ? acc11 : acc10) : (nf ? acc01 : acc00);
      int col = bn0 + wn + nf * 16 + (lane & 15);
      if (col < Ncols) {
#pragma unroll
        for (int r = 0; r < 8; ++r) {
          int gm = bm0 + wm + mf * 16 + r + hi * 8;
          float v = acc[r];
          if (bias) v += bias[gm];
          if (av)   v += (float)av[(size_t)gm * ldadd + col];
          if (oF) oF[(size_t)gm * ldc + col] = v;
          if (oH) oH[(size_t)gm * ldc + col] = (__bf16)v;
        }
      }
    }
  }
}

// ---------------------------------------------------------------------------
__global__ void f32_to_bf16_kernel(const float* __restrict__ in,
                                   __bf16* __restrict__ out, int count) {
  int i = blockIdx.x * blockDim.x + threadIdx.x;
  if (i < count) out[i] = (__bf16)in[i];
}

// q_land / k_land: mean over 64 rows of q/k (stride-768 view), / 64 again.
__global__ __launch_bounds__(256) void landmark_kernel(const __bf16* __restrict__ qkv,
                                                       __bf16* __restrict__ qland,
                                                       __bf16* __restrict__ kland) {
  int e     = threadIdx.x;   // 0..255
  int l     = blockIdx.x;    // 0..63
  int which = blockIdx.y;    // 0 = q, 1 = k
  int n     = blockIdx.z;
  const __bf16* base = qkv + (size_t)n * 768 * 4096 + (size_t)which * 256 + e;
  float s = 0.f;
  for (int i = 0; i < 64; ++i) s += (float)base[(size_t)(l * 64 + i) * 768];
  __bf16* dst = (which ? kland : qland) + (size_t)n * 64 * 256 + l * 256 + e;
  *dst = (__bf16)(s * (1.0f / 4096.0f));
}

// softmax along a 4096-wide row -> bf16
__global__ __launch_bounds__(256) void softmax_row_kernel(const float* __restrict__ S,
                                                          __bf16* __restrict__ out, int cols) {
  const size_t base = ((size_t)blockIdx.z * gridDim.x + blockIdx.x) * cols;
  const float* row = S + base;
  __bf16* orow = out + base;
  __shared__ float red[256];
  int t = threadIdx.x;
  float m = -1e30f;
  for (int i = t; i < cols; i += 256) m = fmaxf(m, row[i]);
  red[t] = m; __syncthreads();
  for (int s = 128; s > 0; s >>= 1) { if (t < s) red[t] = fmaxf(red[t], red[t + s]); __syncthreads(); }
  m = red[0]; __syncthreads();
  float sum = 0.f;
  for (int i = t; i < cols; i += 256) sum += __expf(row[i] - m);
  red[t] = sum; __syncthreads();
  for (int s = 128; s > 0; s >>= 1) { if (t < s) red[t] += red[t + s]; __syncthreads(); }
  float inv = 1.0f / red[0];
  for (int i = t; i < cols; i += 256) orow[i] = (__bf16)(__expf(row[i] - m) * inv);
}

// softmax down a column of `rows` values (stride ld); f32 in place, optional bf16 copy
__global__ void softmax_col_kernel(float* __restrict__ S, __bf16* __restrict__ out,
                                   int rows, int cols, int ld) {
  int j = blockIdx.x * blockDim.x + threadIdx.x;
  if (j >= cols) return;
  size_t base = (size_t)blockIdx.z * rows * ld + j;
  float m = -1e30f;
  for (int i = 0; i < rows; ++i) m = fmaxf(m, S[base + (size_t)i * ld]);
  float sum = 0.f;
  for (int i = 0; i < rows; ++i) sum += __expf(S[base + (size_t)i * ld] - m);
  float inv = 1.0f / sum;
  for (int i = 0; i < rows; ++i) {
    float v = __expf(S[base + (size_t)i * ld] - m) * inv;
    S[base + (size_t)i * ld] = v;
    if (out) out[base + (size_t)i * ld] = (__bf16)v;
  }
}

// Newton-Schulz pseudo-inverse of 64x64 matrices, all in LDS (4 x 16KB).
__global__ __launch_bounds__(256) void newton_inv_kernel(const float* __restrict__ K2,
                                                         __bf16* __restrict__ Vinv) {
  __shared__ float Ks[4096], Vs[4096], Ps[4096], Ts[4096];
  const int t = threadIdx.x;
  const int n = blockIdx.x;
  const float* src = K2 + (size_t)n * 4096;
  for (int i = t; i < 4096; i += 256) Ks[i] = src[i];
  __syncthreads();
  if (t < 64) {
    float cs = 0.f, rs = 0.f;
    for (int i = 0; i < 64; ++i) { cs += fabsf(Ks[i * 64 + t]); rs += fabsf(Ks[t * 64 + i]); }
    Ts[t] = cs; Ts[64 + t] = rs;
  }
  __syncthreads();
  if (t == 0) {
    float v0 = 0.f, vi = 0.f;
    for (int i = 0; i < 64; ++i) { v0 = fmaxf(v0, Ts[i]); vi = fmaxf(vi, Ts[64 + i]); }
    Ts[128] = 1.0f / (v0 * vi);
  }
  __syncthreads();
  float invs = Ts[128];
  for (int i = t; i < 4096; i += 256) {
    int r = i >> 6, c = i & 63;
    Vs[i] = Ks[c * 64 + r] * invs;   // V = K^T / (V0*VI)
  }
  __syncthreads();

  const int row = t >> 2;
  const int jb  = (t & 3) << 4;
  float reg[16];
  for (int iter = 0; iter < 6; ++iter) {
    // P = K @ V
#pragma unroll
    for (int j = 0; j < 16; ++j) reg[j] = 0.f;
    for (int k = 0; k < 64; ++k) {
      float a = Ks[row * 64 + k];
      const float* br = &Vs[k * 64 + jb];
#pragma unroll
      for (int j = 0; j < 16; ++j) reg[j] += a * br[j];
    }
    for (int j = 0; j < 16; ++j) Ps[row * 64 + jb + j] = reg[j];
    __syncthreads();
    // T = 15I - 7P + P@P
#pragma unroll
    for (int j = 0; j < 16; ++j) reg[j] = 0.f;
    for (int k = 0; k < 64; ++k) {
      float a = Ps[row * 64 + k];
      const float* br = &Ps[k * 64 + jb];
#pragma unroll
      for (int j = 0; j < 16; ++j) reg[j] += a * br[j];
    }
    for (int j = 0; j < 16; ++j) {
      int c = jb + j;
      float v = reg[j] - 7.f * Ps[row * 64 + c];
      if (c == row) v += 15.f;
      Ts[row * 64 + c] = v;
    }
    __syncthreads();
    // S = 13I - P@T   (registers, then overwrite P)
#pragma unroll
    for (int j = 0; j < 16; ++j) reg[j] = 0.f;
    for (int k = 0; k < 64; ++k) {
      float a = Ps[row * 64 + k];
      const float* br = &Ts[k * 64 + jb];
#pragma unroll
      for (int j = 0; j < 16; ++j) reg[j] += a * br[j];
    }
    for (int j = 0; j < 16; ++j) {
      int c = jb + j;
      reg[j] = (c == row ? 13.f : 0.f) - reg[j];
    }
    __syncthreads();
    for (int j = 0; j < 16; ++j) Ps[row * 64 + jb + j] = reg[j];
    __syncthreads();
    // V = 0.25 * V @ S
#pragma unroll
    for (int j = 0; j < 16; ++j) reg[j] = 0.f;
    for (int k = 0; k < 64; ++k) {
      float a = Vs[row * 64 + k];
      const float* br = &Ps[k * 64 + jb];
#pragma unroll
      for (int j = 0; j < 16; ++j) reg[j] += a * br[j];
    }
    __syncthreads();
    for (int j = 0; j < 16; ++j) Vs[row * 64 + jb + j] = 0.25f * reg[j];
    __syncthreads();
  }
  __bf16* dst = Vinv + (size_t)n * 4096;
  for (int i = t; i < 4096; i += 256) dst[i] = (__bf16)Vs[i];
}

// ---------------------------------------------------------------------------
extern "C" void kernel_launch(void* const* d_in, const int* in_sizes, int n_in,
                              void* d_out, int out_size, void* d_ws, size_t ws_size,
                              hipStream_t stream) {
  const float* x     = (const float*)d_in[0];   // [8,256,64,64]
  const float* w_qkv = (const float*)d_in[1];   // [768,256]
  const float* b_qkv = (const float*)d_in[2];   // [768]
  const float* w_out = (const float*)d_in[3];   // [256,256]
  const float* b_out = (const float*)d_in[4];   // [256]
  float* out = (float*)d_out;                   // [8,256,64,64]
  char*  ws  = (char*)d_ws;

  // workspace layout (bytes, all 256-aligned), total ~103 MB
  __bf16* xb     = (__bf16*)(ws + 0);          // 8*256*4096 bf16
  __bf16* wqkvb  = (__bf16*)(ws + 16777216);   // 768*256
  __bf16* woutb  = (__bf16*)(ws + 17170432);   // 256*256
  __bf16* qkvb   = (__bf16*)(ws + 17301504);   // 8*768*4096  (view: [4096,768] -> q|k|v)
  __bf16* qlandb = (__bf16*)(ws + 67633152);   // 8*64*256
  __bf16* klandb = (__bf16*)(ws + 67895296);   // 8*64*256
  float*  Sf     = (float*) (ws + 68157440);   // 8*64*4096 f32 (reused S1t then S3)
  __bf16* k1tb   = (__bf16*)(ws + 76546048);   // 8*64*4096
  __bf16* k3b    = (__bf16*)(ws + 80740352);   // 8*64*4096
  float*  S2f    = (float*) (ws + 84934656);   // 8*64*64 f32
  __bf16* k2invb = (__bf16*)(ws + 85065728);   // 8*64*64
  __bf16* T1b    = (__bf16*)(ws + 85131264);   // 8*64*256
  __bf16* T2b    = (__bf16*)(ws + 85393408);   // 8*64*256
  __bf16* outb   = (__bf16*)(ws + 85655552);   // 8*4096*256 (view: [256,4096])

  f32_to_bf16_kernel<<<(8388608 + 255) / 256, 256, 0, stream>>>(x, xb, 8388608);
  f32_to_bf16_kernel<<<(196608  + 255) / 256, 256, 0, stream>>>(w_qkv, wqkvb, 196608);
  f32_to_bf16_kernel<<<(65536   + 255) / 256, 256, 0, stream>>>(w_out, woutb, 65536);

  // QKV[768,4096] = w_qkv @ x + b_qkv          (per batch)
  gemm_bf16_wmma<<<dim3(32, 12, 8), 256, 0, stream>>>(
      wqkvb, 256, 0LL, 0,
      xb, 4096, 1048576LL, 0,
      nullptr, qkvb, 4096, 3145728LL,
      b_qkv, nullptr, 0, 0LL, 256, 4096);

  landmark_kernel<<<dim3(64, 2, 8), 256, 0, stream>>>(qkvb, qlandb, klandb);

  // S1t[64,4096] = k_land @ q^T   (softmax over HW = rows of S1t)
  gemm_bf16_wmma<<<dim3(32, 1, 8), 256, 0, stream>>>(
      klandb, 256, 16384LL, 0,
      qkvb, 768, 3145728LL, 1,
      Sf, nullptr, 4096, 262144LL,
      nullptr, nullptr, 0, 0LL, 256, 4096);
  softmax_row_kernel<<<dim3(64, 1, 8), 256, 0, stream>>>(Sf, k1tb, 4096);

  // S3[64,4096] = q_land @ k^T ; softmax over the 64-dim (columns)
  gemm_bf16_wmma<<<dim3(32, 1, 8), 256, 0, stream>>>(
      qlandb, 256, 16384LL, 0,
      qkvb + 256, 768, 3145728LL, 1,
      Sf, nullptr, 4096, 262144LL,
      nullptr, nullptr, 0, 0LL, 256, 4096);
  softmax_col_kernel<<<dim3(16, 1, 8), 256, 0, stream>>>(Sf, k3b, 64, 4096, 4096);

  // S2[64,64] = q_land @ k_land^T ; softmax over rows ; Newton-Schulz inverse
  gemm_bf16_wmma<<<dim3(1, 1, 8), 256, 0, stream>>>(
      qlandb, 256, 16384LL, 0,
      klandb, 256, 16384LL, 1,
      S2f, nullptr, 64, 4096LL,
      nullptr, nullptr, 0, 0LL, 256, 64);
  softmax_col_kernel<<<dim3(1, 1, 8), 256, 0, stream>>>(S2f, nullptr, 64, 64, 64);
  newton_inv_kernel<<<8, 256, 0, stream>>>(S2f, k2invb);

  // T1[64,256] = k3 @ v
  gemm_bf16_wmma<<<dim3(2, 1, 8), 256, 0, stream>>>(
      k3b, 4096, 262144LL, 0,
      qkvb + 512, 768, 3145728LL, 0,
      nullptr, T1b, 256, 16384LL,
      nullptr, nullptr, 0, 0LL, 4096, 256);

  // T2[64,256] = k2_inv @ T1
  gemm_bf16_wmma<<<dim3(2, 1, 8), 256, 0, stream>>>(
      k2invb, 64, 4096LL, 0,
      T1b, 256, 16384LL, 0,
      nullptr, T2b, 256, 16384LL,
      nullptr, nullptr, 0, 0LL, 64, 256);

  // out[4096,256] = k1 @ T2 + v    (k1 via transA from k1t)
  gemm_bf16_wmma<<<dim3(2, 64, 8), 256, 0, stream>>>(
      k1tb, 4096, 262144LL, 1,
      T2b, 256, 16384LL, 0,
      nullptr, outb, 256, 1048576LL,
      nullptr, qkvb + 512, 768, 3145728LL, 64, 256);

  // final[256,4096] = w_out @ out_img + b_out   (out buffer viewed [256,4096])
  gemm_bf16_wmma<<<dim3(32, 4, 8), 256, 0, stream>>>(
      woutb, 256, 0LL, 0,
      outb, 4096, 1048576LL, 0,
      out, nullptr, 4096, 1048576LL,
      b_out, nullptr, 0, 0LL, 256, 4096);

  (void)in_sizes; (void)n_in; (void)out_size; (void)ws_size;
}
